// EdgeEnhancedGraphTransformer3D_17721035063342
// MI455X (gfx1250) — matrix operand order repaired and checked by reading, not compiled, not run on hardware
//
#include <hip/hip_runtime.h>
#include <hip/hip_bf16.h>

// Problem constants (from reference)
#define NN   20000
#define NE   320000
#define IND  48
#define IED  4
#define HDIM 128
#define NHEAD 8
#define HDSZ 16
#define NLAY 6
#define NBATCH 64
#define FFD  512

typedef __attribute__((ext_vector_type(16))) __bf16 v16bf;
typedef __attribute__((ext_vector_type(8)))  __bf16 v8bf;
typedef __attribute__((ext_vector_type(8)))  float  v8f;

// ---------------------------------------------------------------- utilities
__device__ __forceinline__ float wave_reduce_sum(float v) {
  #pragma unroll
  for (int m = 16; m >= 1; m >>= 1) v += __shfl_xor(v, m, 32);
  return v;
}

// order-preserving float<->uint encoding for atomic segment-max
__device__ __forceinline__ unsigned enc_f32(float f) {
  unsigned u = __float_as_uint(f);
  return (u & 0x80000000u) ? ~u : (u | 0x80000000u);
}
__device__ __forceinline__ float dec_f32(unsigned u) {
  unsigned b = (u & 0x80000000u) ? (u & 0x7FFFFFFFu) : ~u;
  return __uint_as_float(b);
}

__device__ __forceinline__ void atomic_add_f32(float* p, float v) {
  __hip_atomic_fetch_add(p, v, __ATOMIC_RELAXED, __HIP_MEMORY_SCOPE_AGENT);
}

__device__ __forceinline__ v16bf combine16(v8bf lo, v8bf hi) {
  return __builtin_shufflevector(lo, hi, 0,1,2,3,4,5,6,7,8,9,10,11,12,13,14,15);
}

// ---------------------------------------------------------------- node embed
__global__ void node_embed_kernel(const float* __restrict__ x,
                                  const float* __restrict__ W,
                                  const float* __restrict__ b,
                                  float* __restrict__ h) {
  __shared__ float xs[IND];
  int n = blockIdx.x, t = threadIdx.x;
  if (t < IND) xs[t] = x[(size_t)n * IND + t];
  __syncthreads();
  float acc = b[t];
  #pragma unroll
  for (int k = 0; k < IND; ++k) acc += xs[k] * W[k * HDIM + t];
  h[(size_t)n * HDIM + t] = acc;
}

// ---------------------------------------------------------------- layernorm
// wave32 per row, 4 floats/lane; writes bf16 copy (GEMM input) and/or f32
__global__ void layernorm_kernel(const float* __restrict__ in,
                                 const float* __restrict__ s,
                                 const float* __restrict__ b,
                                 __bf16* __restrict__ out_bf,
                                 float* __restrict__ out_f32,
                                 int nrows) {
  int wave = threadIdx.x >> 5, lane = threadIdx.x & 31;
  int row = blockIdx.x * 8 + wave;
  if (row >= nrows) return;
  float4 v = ((const float4*)(in + (size_t)row * HDIM))[lane];
  float mean = wave_reduce_sum(v.x + v.y + v.z + v.w) * (1.0f / HDIM);
  float dx = v.x - mean, dy = v.y - mean, dz = v.z - mean, dw = v.w - mean;
  float var = wave_reduce_sum(dx*dx + dy*dy + dz*dz + dw*dw) * (1.0f / HDIM);
  float rstd = rsqrtf(var + 1e-5f);
  int c = lane * 4;
  float o0 = dx * rstd * s[c+0] + b[c+0];
  float o1 = dy * rstd * s[c+1] + b[c+1];
  float o2 = dz * rstd * s[c+2] + b[c+2];
  float o3 = dw * rstd * s[c+3] + b[c+3];
  if (out_bf) {
    __bf16* o = out_bf + (size_t)row * HDIM + c;
    o[0] = (__bf16)o0; o[1] = (__bf16)o1; o[2] = (__bf16)o2; o[3] = (__bf16)o3;
  }
  if (out_f32) {
    float4 ov = {o0, o1, o2, o3};
    ((float4*)(out_f32 + (size_t)row * HDIM))[lane] = ov;
  }
}

// ---------------------------------------------------------------- f32 -> bf16
__global__ void f2bf_kernel(const float* __restrict__ in, __bf16* __restrict__ out, int n) {
  int t = blockIdx.x * blockDim.x + threadIdx.x;
  if (t < n) out[t] = (__bf16)in[t];
}

// ---------------------------------------------------------------- weight pack
// Pack fp32 W[K,N] (row-major) into bf16 B-fragment layout:
// fragment (kt, ntg): 32 lanes x 16 bf16 contiguous per lane.
// lane: n = ntg*16 + (lane&15); K base = (lane>=16)?16:0; element e -> K = kt*32 + base + e
__global__ void pack_w_kernel(const float* __restrict__ W, __bf16* __restrict__ P,
                              int K, int N) {
  int t = blockIdx.x * blockDim.x + threadIdx.x;
  if (t >= K * N) return;
  int e    = t & 15;
  int lane = (t >> 4) & 31;
  int frag = t >> 9;
  int nfrag = N >> 4;
  int ntg = frag % nfrag;
  int kt  = frag / nfrag;
  int k = kt * 32 + ((lane >= 16) ? 16 : 0) + e;
  int n = ntg * 16 + (lane & 15);
  P[t] = (__bf16)W[(size_t)k * N + n];
}

// ---------------------------------------------------------------- WMMA GEMM
// C[M,NTOT] = A[M,KTOT](bf16) @ Wpacked + bias (+res) (+relu).
// Block = 8 waves x 16 rows over a 128-col slab (blockIdx.y).
// B (weights) are staged through LDS, double-buffered: all 8 waves share one
// 8KB slab per K-step instead of 8x redundant global streams. Software
// pipeline: issue global loads for step kt+1 -> 8 WMMAs from LDS (DScnt-only
// waits) -> ds_store stage(kt+1) (LOADcnt wait lands after the WMMA burst).
template<bool RELU, bool RES, bool WF32, bool WBF16, int NTOT, int KTOT>
__global__ __launch_bounds__(256)
void gemm_wmma_kernel(const __bf16* __restrict__ A,
                      const __bf16* __restrict__ P,
                      const float* __restrict__ bias,
                      const float* __restrict__ res,
                      float* __restrict__ C,
                      __bf16* __restrict__ Cbf,
                      int M) {
  constexpr int NFRAG  = NTOT / 16;   // B fragments per K-step over full N
  constexpr int KSTEPS = KTOT / 32;
  // [buffer][fragment][lane][elem] : 2 x 8 x 32 x 16 bf16 = 16KB
  __shared__ alignas(16) __bf16 ldsb[2][8][32][16];

  int wave = threadIdx.x >> 5, lane = threadIdx.x & 31;
  int mbase = blockIdx.x * 128 + wave * 16;
  int c0    = blockIdx.y * 128;
  int ntg0  = c0 >> 4;
  int kbase = (lane >= 16) ? 16 : 0;         // A-fragment lane swizzle
  int mrow  = mbase + (lane & 15);

  const __bf16* abase = A + (size_t)((mrow < M) ? mrow : 0) * KTOT + kbase;

  // stage K-step 0 into LDS buffer 0 (8KB = 512 uint4, 2 per thread)
  {
    const uint4* src = (const uint4*)(P + (size_t)ntg0 * 512);
    uint4* dst = (uint4*)(&ldsb[0][0][0][0]);
    dst[threadIdx.x]       = src[threadIdx.x];
    dst[threadIdx.x + 256] = src[threadIdx.x + 256];
  }
  v16bf a = combine16(*(const v8bf*)(abase), *(const v8bf*)(abase + 8));
  __syncthreads();

  v8f acc[8] = {};

  #pragma unroll
  for (int kt = 0; kt < KSTEPS; ++kt) {
    v16bf an;
    uint4 st0, st1;
    if (kt + 1 < KSTEPS) {
      // issue next-step global loads first so they overlap the WMMA burst
      const uint4* src = (const uint4*)(P + (size_t)((kt + 1) * NFRAG + ntg0) * 512);
      st0 = src[threadIdx.x];
      st1 = src[threadIdx.x + 256];
      const __bf16* ap = abase + (kt + 1) * 32;
      an = combine16(*(const v8bf*)(ap), *(const v8bf*)(ap + 8));
    }
    #pragma unroll
    for (int nt = 0; nt < 8; ++nt) {
      const __bf16* bp = &ldsb[kt & 1][nt][lane][0];
      v16bf bfr = combine16(*(const v8bf*)(bp), *(const v8bf*)(bp + 8));
      acc[nt] = __builtin_amdgcn_wmma_f32_16x16x32_bf16(
          false, a, false, bfr, (short)0, acc[nt], false, false);
    }
    if (kt + 1 < KSTEPS) {
      uint4* dst = (uint4*)(&ldsb[(kt + 1) & 1][0][0][0]);
      dst[threadIdx.x]       = st0;     // waits LOADcnt here (after WMMAs)
      dst[threadIdx.x + 256] = st1;
      a = an;
    }
    __syncthreads();
  }

  #pragma unroll
  for (int nt = 0; nt < 8; ++nt) {
    int col = c0 + nt * 16 + (lane & 15);
    float bv = bias ? bias[col] : 0.0f;
    #pragma unroll
    for (int r = 0; r < 8; ++r) {
      int row = mbase + ((lane < 16) ? r : (8 + r));
      if (row < M) {
        float v = acc[nt][r] + bv;
        if (RES)  v += res[(size_t)row * NTOT + col];
        if (RELU) v = fmaxf(v, 0.0f);
        if (WF32)  C[(size_t)row * NTOT + col] = v;
        if (WBF16) Cbf[(size_t)row * NTOT + col] = (__bf16)v;
      }
    }
  }
}

// ---------------------------------------------------------------- score consts
// Fold edge_emb@We through tiny edge_attr:  M_l = edge_W(4x128) @ We[l](128x8),
// c_l = edge_b@We[l] + be[l]; plus HD-summed Wr/br.  72 floats per layer.
__global__ void score_consts_kernel(const float* __restrict__ edge_W,
                                    const float* __restrict__ edge_b,
                                    const float* __restrict__ We,
                                    const float* __restrict__ be,
                                    const float* __restrict__ Wr,
                                    const float* __restrict__ br,
                                    float* __restrict__ out) {
  int t = threadIdx.x;
  if (t >= NLAY * NHEAD) return;
  int l = t / NHEAD, hh = t % NHEAD;
  float* Ml  = out + l * 72;
  float* cl  = out + l * 72 + 32;
  float* Wrs = out + l * 72 + 40;
  float* brs = out + l * 72 + 64;
  const float* Wel = We + (size_t)l * HDIM * NHEAD;
  for (int j = 0; j < IED; ++j) {
    float s = 0.f;
    for (int k = 0; k < HDIM; ++k) s += edge_W[j * HDIM + k] * Wel[k * NHEAD + hh];
    Ml[j * NHEAD + hh] = s;
  }
  float c = be[l * NHEAD + hh];
  for (int k = 0; k < HDIM; ++k) c += edge_b[k] * Wel[k * NHEAD + hh];
  cl[hh] = c;
  for (int j = 0; j < 3; ++j) {
    float s = 0.f;
    for (int d = 0; d < HDSZ; ++d) s += Wr[(size_t)l * 3 * HDIM + j * HDIM + hh * HDSZ + d];
    Wrs[j * NHEAD + hh] = s;
  }
  float s = 0.f;
  for (int d = 0; d < HDSZ; ++d) s += br[(size_t)l * HDIM + hh * HDSZ + d];
  brs[hh] = s;
}

// ---------------------------------------------------------------- edge scores
// wave per edge; lane = 4 dims, head = lane>>2; atomic segment max.
__global__ void score_kernel(const float* __restrict__ q, const float* __restrict__ k,
                             const int* __restrict__ row, const int* __restrict__ col,
                             const float* __restrict__ edge_attr,
                             const float* __restrict__ pos,
                             const float* __restrict__ cst,
                             float* __restrict__ sc, unsigned* __restrict__ mxu) {
  int wave = threadIdx.x >> 5, lane = threadIdx.x & 31;
  int e = blockIdx.x * 8 + wave;
  if (e >= NE) return;
  int r = row[e], c = col[e];
  float4 qv = ((const float4*)(q + (size_t)r * HDIM))[lane];
  float4 kv = ((const float4*)(k + (size_t)c * HDIM))[lane];
  float dot = qv.x * kv.x + qv.y * kv.y + qv.z * kv.z + qv.w * kv.w;
  dot += __shfl_xor(dot, 1, 32);
  dot += __shfl_xor(dot, 2, 32);
  if ((lane & 3) == 0) {
    int hh = lane >> 2;
    float s = dot * 0.25f;                         // 1/sqrt(HD), HD=16
    const float* Ml  = cst;
    const float* cl  = cst + 32;
    const float* Wrs = cst + 40;
    const float* brs = cst + 64;
    float es = cl[hh];
    #pragma unroll
    for (int j = 0; j < IED; ++j) es += edge_attr[(size_t)e * IED + j] * Ml[j * NHEAD + hh];
    float rp0 = pos[(size_t)r*3+0] - pos[(size_t)c*3+0];
    float rp1 = pos[(size_t)r*3+1] - pos[(size_t)c*3+1];
    float rp2 = pos[(size_t)r*3+2] - pos[(size_t)c*3+2];
    s += es + brs[hh] + rp0*Wrs[hh] + rp1*Wrs[NHEAD+hh] + rp2*Wrs[2*NHEAD+hh];
    sc[(size_t)e * NHEAD + hh] = s;
    atomicMax(&mxu[(size_t)r * NHEAD + hh], enc_f32(s));
  }
}

// ---------------------------------------------------------------- softmax exp
__global__ void exp_kernel(const int* __restrict__ row, float* __restrict__ sc,
                           const unsigned* __restrict__ mxu, float* __restrict__ den) {
  int t = blockIdx.x * blockDim.x + threadIdx.x;
  if (t >= NE * NHEAD) return;
  int e = t >> 3, hh = t & 7;
  int r = row[e];
  float ex = __expf(sc[t] - dec_f32(mxu[(size_t)r * NHEAD + hh]));
  sc[t] = ex;
  atomic_add_f32(&den[(size_t)r * NHEAD + hh], ex);
}

// ---------------------------------------------------------------- aggregation
// wave per edge; lane handles dims [lane*4, +4), head = lane>>2; atomic scatter.
__global__ void agg_kernel(const int* __restrict__ row, const int* __restrict__ col,
                           const float* __restrict__ sc, const float* __restrict__ den,
                           const float* __restrict__ v, float* __restrict__ agg) {
  int wave = threadIdx.x >> 5, lane = threadIdx.x & 31;
  int e = blockIdx.x * 8 + wave;
  if (e >= NE) return;
  int r = row[e], c = col[e];
  int hh = lane >> 2;
  float p = sc[(size_t)e * NHEAD + hh] / (den[(size_t)r * NHEAD + hh] + 1e-10f);
  float4 vv = ((const float4*)(v + (size_t)c * HDIM))[lane];
  float* ap = agg + (size_t)r * HDIM + lane * 4;
  atomic_add_f32(ap + 0, p * vv.x);
  atomic_add_f32(ap + 1, p * vv.y);
  atomic_add_f32(ap + 2, p * vv.z);
  atomic_add_f32(ap + 3, p * vv.w);
}

// ---------------------------------------------------------------- pooling
__global__ void pool_kernel(const float* __restrict__ h, const int* __restrict__ batch,
                            float* __restrict__ gsum, float* __restrict__ cnt) {
  int t = blockIdx.x * blockDim.x + threadIdx.x;
  if (t >= NN * HDIM) return;
  int n = t >> 7, d = t & 127;
  int b = batch[n];
  atomic_add_f32(&gsum[(size_t)b * HDIM + d], h[t]);
  if (d == 0) atomic_add_f32(&cnt[b], 1.0f);
}
__global__ void pool_div_kernel(const float* __restrict__ gsum, const float* __restrict__ cnt,
                                float* __restrict__ out) {
  int t = blockIdx.x * blockDim.x + threadIdx.x;
  if (t >= NBATCH * HDIM) return;
  out[t] = gsum[t] / fmaxf(cnt[t >> 7], 1.0f);
}

// ================================================================ launch
extern "C" void kernel_launch(void* const* d_in, const int* in_sizes, int n_in,
                              void* d_out, int out_size, void* d_ws, size_t ws_size,
                              hipStream_t stream) {
  (void)in_sizes; (void)n_in; (void)out_size; (void)ws_size;
  const float* x        = (const float*)d_in[0];
  const float* pos      = (const float*)d_in[1];
  const float* edge_attr= (const float*)d_in[2];
  const int*   eidx     = (const int*)  d_in[3];
  const int*   batch    = (const int*)  d_in[4];
  const float* node_W   = (const float*)d_in[5];
  const float* node_b   = (const float*)d_in[6];
  const float* edge_W   = (const float*)d_in[7];
  const float* edge_b   = (const float*)d_in[8];
  const float* ln1_s    = (const float*)d_in[9];
  const float* ln1_b    = (const float*)d_in[10];
  const float* Wq       = (const float*)d_in[11];
  const float* bq       = (const float*)d_in[12];
  const float* Wk       = (const float*)d_in[13];
  const float* bk       = (const float*)d_in[14];
  const float* Wv       = (const float*)d_in[15];
  const float* bv       = (const float*)d_in[16];
  const float* Wo       = (const float*)d_in[17];
  const float* bo       = (const float*)d_in[18];
  const float* We       = (const float*)d_in[19];
  const float* be       = (const float*)d_in[20];
  const float* Wr       = (const float*)d_in[21];
  const float* br       = (const float*)d_in[22];
  const float* ln2_s    = (const float*)d_in[23];
  const float* ln2_b    = (const float*)d_in[24];
  const float* W1       = (const float*)d_in[25];
  const float* b1       = (const float*)d_in[26];
  const float* W2       = (const float*)d_in[27];
  const float* b2       = (const float*)d_in[28];
  const float* fn_s     = (const float*)d_in[29];
  const float* fn_b     = (const float*)d_in[30];

  const int* row = eidx;
  const int* col = eidx + NE;

  // ---- workspace carve (256B aligned)
  char* w = (char*)d_ws;
  auto carve = [&](size_t bytes) { char* p = w; w += (bytes + 255) & ~(size_t)255; return p; };
  float*    h    = (float*)   carve((size_t)NN * HDIM * 4);
  __bf16*   hnb  = (__bf16*)  carve((size_t)NN * HDIM * 2);
  float*    q    = (float*)   carve((size_t)NN * HDIM * 4);
  float*    kb   = (float*)   carve((size_t)NN * HDIM * 4);
  float*    vb   = (float*)   carve((size_t)NN * HDIM * 4);
  float*    sc   = (float*)   carve((size_t)NE * NHEAD * 4);
  unsigned* mxu  = (unsigned*)carve((size_t)NN * NHEAD * 4);
  float*    den  = (float*)   carve((size_t)NN * NHEAD * 4);
  float*    agg  = (float*)   carve((size_t)NN * HDIM * 4);
  __bf16*   aggb = (__bf16*)  carve((size_t)NN * HDIM * 2);
  __bf16*   ffb  = (__bf16*)  carve((size_t)NN * FFD * 2);
  __bf16*   pWq  = (__bf16*)  carve((size_t)NLAY * HDIM * HDIM * 2);
  __bf16*   pWk  = (__bf16*)  carve((size_t)NLAY * HDIM * HDIM * 2);
  __bf16*   pWv  = (__bf16*)  carve((size_t)NLAY * HDIM * HDIM * 2);
  __bf16*   pWo  = (__bf16*)  carve((size_t)NLAY * HDIM * HDIM * 2);
  __bf16*   pW1  = (__bf16*)  carve((size_t)NLAY * HDIM * FFD * 2);
  __bf16*   pW2  = (__bf16*)  carve((size_t)NLAY * FFD * HDIM * 2);
  float*    cst  = (float*)   carve((size_t)NLAY * 72 * 4);
  float*    gsum = (float*)   carve((size_t)NBATCH * HDIM * 4);
  float*    cnt  = (float*)   carve((size_t)NBATCH * 4);

  const int TPB = 256;
  const int MB = (NN + 127) / 128;          // GEMM M tiles
  const int HH2 = HDIM * HDIM;              // 16384
  const int HF  = HDIM * FFD;               // 65536

  // ---- pack weights to bf16 B-fragment layout
  for (int l = 0; l < NLAY; ++l) {
    pack_w_kernel<<<HH2 / TPB, TPB, 0, stream>>>(Wq + (size_t)l*HH2, pWq + (size_t)l*HH2, HDIM, HDIM);
    pack_w_kernel<<<HH2 / TPB, TPB, 0, stream>>>(Wk + (size_t)l*HH2, pWk + (size_t)l*HH2, HDIM, HDIM);
    pack_w_kernel<<<HH2 / TPB, TPB, 0, stream>>>(Wv + (size_t)l*HH2, pWv + (size_t)l*HH2, HDIM, HDIM);
    pack_w_kernel<<<HH2 / TPB, TPB, 0, stream>>>(Wo + (size_t)l*HH2, pWo + (size_t)l*HH2, HDIM, HDIM);
    pack_w_kernel<<<HF  / TPB, TPB, 0, stream>>>(W1 + (size_t)l*HF,  pW1 + (size_t)l*HF,  HDIM, FFD);
    pack_w_kernel<<<HF  / TPB, TPB, 0, stream>>>(W2 + (size_t)l*HF,  pW2 + (size_t)l*HF,  FFD, HDIM);
  }
  score_consts_kernel<<<1, 64, 0, stream>>>(edge_W, edge_b, We, be, Wr, br, cst);

  // ---- node embedding
  node_embed_kernel<<<NN, HDIM, 0, stream>>>(x, node_W, node_b, h);

  const int LNB = (NN + 7) / 8;             // layernorm blocks (8 rows/block)

  for (int l = 0; l < NLAY; ++l) {
    // LN1 -> bf16
    layernorm_kernel<<<LNB, TPB, 0, stream>>>(h, ln1_s + l*HDIM, ln1_b + l*HDIM, hnb, nullptr, NN);
    // Q, K, V projections (WMMA)
    gemm_wmma_kernel<false,false,true,false,HDIM,HDIM><<<dim3(MB,1), TPB, 0, stream>>>(
        hnb, pWq + (size_t)l*HH2, bq + l*HDIM, nullptr, q,  nullptr, NN);
    gemm_wmma_kernel<false,false,true,false,HDIM,HDIM><<<dim3(MB,1), TPB, 0, stream>>>(
        hnb, pWk + (size_t)l*HH2, bk + l*HDIM, nullptr, kb, nullptr, NN);
    gemm_wmma_kernel<false,false,true,false,HDIM,HDIM><<<dim3(MB,1), TPB, 0, stream>>>(
        hnb, pWv + (size_t)l*HH2, bv + l*HDIM, nullptr, vb, nullptr, NN);
    // segmented softmax state
    hipMemsetAsync(mxu, 0, (size_t)NN * NHEAD * 4, stream);
    hipMemsetAsync(den, 0, (size_t)NN * NHEAD * 4, stream);
    hipMemsetAsync(agg, 0, (size_t)NN * HDIM * 4, stream);
    score_kernel<<<NE / 8, TPB, 0, stream>>>(q, kb, row, col, edge_attr, pos, cst + l*72, sc, mxu);
    exp_kernel<<<(NE * NHEAD) / TPB, TPB, 0, stream>>>(row, sc, mxu, den);
    agg_kernel<<<NE / 8, TPB, 0, stream>>>(row, col, sc, den, vb, agg);
    // O projection + residual
    f2bf_kernel<<<(NN * HDIM + TPB - 1) / TPB, TPB, 0, stream>>>(agg, aggb, NN * HDIM);
    gemm_wmma_kernel<false,true,true,false,HDIM,HDIM><<<dim3(MB,1), TPB, 0, stream>>>(
        aggb, pWo + (size_t)l*HH2, bo + l*HDIM, h, h, nullptr, NN);
    // FFN
    layernorm_kernel<<<LNB, TPB, 0, stream>>>(h, ln2_s + l*HDIM, ln2_b + l*HDIM, hnb, nullptr, NN);
    gemm_wmma_kernel<true,false,false,true,FFD,HDIM><<<dim3(MB,4), TPB, 0, stream>>>(
        hnb, pW1 + (size_t)l*HF, b1 + l*FFD, nullptr, nullptr, ffb, NN);
    gemm_wmma_kernel<false,true,true,false,HDIM,FFD><<<dim3(MB,1), TPB, 0, stream>>>(
        ffb, pW2 + (size_t)l*HF, b2 + l*HDIM, h, h, nullptr, NN);
  }

  // final LN straight into d_out, then mean-pool per graph
  float* out_h = (float*)d_out;
  float* out_g = out_h + (size_t)NN * HDIM;
  layernorm_kernel<<<LNB, TPB, 0, stream>>>(h, fn_s, fn_b, nullptr, out_h, NN);
  hipMemsetAsync(gsum, 0, (size_t)NBATCH * HDIM * 4, stream);
  hipMemsetAsync(cnt,  0, (size_t)NBATCH * 4, stream);
  pool_kernel<<<(NN * HDIM) / TPB, TPB, 0, stream>>>(out_h, batch, gsum, cnt);
  pool_div_kernel<<<(NBATCH * HDIM + TPB - 1) / TPB, TPB, 0, stream>>>(gsum, cnt, out_g);
}
